// DisentangledDecoder_41798621724872
// MI455X (gfx1250) — compile-verified, matrix operand order checked by reading, not run on hardware
//
#include <hip/hip_runtime.h>
#include <hip/hip_bf16.h>
#include <stdint.h>

// ---------------------------------------------------------------------------
// MI455X (gfx1250) implementation of the disentangled video decoder.
//  - wave32, WMMA f32_16x16x32_f16 for all large GEMMs (LSTM / FF / FC / deconvs)
//  - LSTM fused into ONE persistent 512-thread workgroup, h ping-pong in LDS
//  - ConvTranspose2d(k4,s2,p1) as implicit GEMM via 4 output-parity classes
//  - fp32 accumulate everywhere, fp32 BN batch statistics, final layer scalar
// ---------------------------------------------------------------------------

typedef __attribute__((ext_vector_type(16))) _Float16 v16h;
typedef __attribute__((ext_vector_type(8)))  _Float16 v8h;
typedef __attribute__((ext_vector_type(8)))  float    v8f;

#define LEAKK 0.2f

__device__ __forceinline__ float sigf(float x)  { return 1.0f / (1.0f + __expf(-x)); }
__device__ __forceinline__ float lrelu(float x) { return x >= 0.f ? x : LEAKK * x; }

__device__ __forceinline__ v8f wmma_f16(v16h a, v16h b, v8f c) {
  return __builtin_amdgcn_wmma_f32_16x16x32_f16(false, a, false, b, (short)0, c, false, false);
}

// A/B fragment load from a row-major f16 matrix (row-stride ld halves).
// Lane l: row = r0 + (l&15); K chunks [k0+8*(l>>4) .. +8) and (+16 .. +24).
__device__ __forceinline__ v16h frag_ld(const _Float16* p, int ld, int r0, int k0) {
  int l = threadIdx.x & 31;
  const _Float16* base = p + (size_t)(r0 + (l & 15)) * ld + (k0 + ((l >> 4) << 3));
  union { v16h v; struct { v8h a, b; } s; } u;
  u.s.a = *(const v8h*)base;
  u.s.b = *(const v8h*)(base + 16);
  return u.v;
}

// --------------------------- prep kernels ----------------------------------

__global__ void k_cvt_lstm(const float* __restrict__ Wih, const float* __restrict__ Whh,
                           _Float16* __restrict__ WhhH, _Float16* __restrict__ WcombH) {
  int i = blockIdx.x * 256 + threadIdx.x;
  if (i >= 2048 * 512) return;
  float a = Wih[i], b = Whh[i];
  WhhH[i]   = (_Float16)b;
  WcombH[i] = (_Float16)(a + b);   // x_t == h_{t-1} for t>=1
}

__global__ void k_cvt(const float* __restrict__ s, _Float16* __restrict__ d, int n) {
  int i = blockIdx.x * 256 + threadIdx.x;
  if (i < n) d[i] = (_Float16)s[i];
}

__global__ void k_bias0(const float* __restrict__ bih, const float* __restrict__ bhh,
                        const float* __restrict__ Wih, const float* __restrict__ init_in,
                        float* __restrict__ bias0, float* __restrict__ biasC) {
  int nn = blockIdx.x * 256 + threadIdx.x;
  if (nn >= 2048) return;
  float s = bih[nn] + bhh[nn];
  biasC[nn] = s;
  const float* wr = Wih + (size_t)nn * 512;
  float d = 0.f;
  for (int k = 0; k < 512; ++k) d += init_in[k] * wr[k];  // x0 @ W_ih.T (batch-invariant)
  bias0[nn] = s + d;
}

__global__ void k_h0c0(const float* __restrict__ zc, const float* __restrict__ zm,
                       const float* __restrict__ Wh, const float* __restrict__ bh,
                       const float* __restrict__ Wc, const float* __restrict__ bc,
                       _Float16* __restrict__ h0, float* __restrict__ c0) {
  int i = blockIdx.x * 256 + threadIdx.x;
  if (i >= 64 * 512) return;
  int bb = i >> 9, hh = i & 511;
  const float* zr0 = zc + bb * 128;
  const float* zr1 = zm + bb * 128;
  const float* w0 = Wh + (size_t)hh * 256;
  const float* w1 = Wc + (size_t)hh * 256;
  float a = bh[hh], c = bc[hh];
  for (int k = 0; k < 128; ++k) {
    float z0 = zr0[k], z1 = zr1[k];
    a += z0 * w0[k] + z1 * w0[128 + k];
    c += z0 * w1[k] + z1 * w1[128 + k];
  }
  h0[i] = (_Float16)a;
  c0[i] = c;
}

// Build per-parity deconv weight matrices: bp[par][co][tap*Cin+ci] =
//   w[ci][co][3-2*ty-py][3-2*tx-px]   (w is PyTorch [in,out,4,4])
__global__ void k_build_bp(const float* __restrict__ w, _Float16* __restrict__ bp,
                           int Cin, int Cout) {
  int total = 16 * Cin * Cout;
  int i = blockIdx.x * 256 + threadIdx.x;
  if (i >= total) return;
  int K = 4 * Cin;
  int co = i % Cout; int r = i / Cout;
  int k  = r % K;    int par = r / K;
  int ci = k % Cin;  int tap = k / Cin;
  int ty = tap >> 1, tx = tap & 1, py = par >> 1, px = par & 1;
  bp[((size_t)(par * Cout + co)) * K + k] =
      (_Float16)w[(((size_t)ci * Cout + co) * 4 + (3 - 2 * ty - py)) * 4 + (3 - 2 * tx - px)];
}

__global__ void k_featzc(const float* __restrict__ zc, _Float16* __restrict__ feat) {
  int i = blockIdx.x * 256 + threadIdx.x;
  if (i >= 1280 * 128) return;
  int r = i >> 7, c = i & 127;
  feat[(size_t)r * 640 + 512 + c] = (_Float16)zc[(r / 20) * 128 + c];
}

__global__ void k_zero_h(_Float16* __restrict__ p, long n) {
  long i = (long)blockIdx.x * blockDim.x + threadIdx.x;
  long stride = (long)gridDim.x * blockDim.x;
  for (; i < n; i += stride) p[i] = (_Float16)0.f;
}

// --------------------------- LSTM (persistent) -----------------------------
// One 512-thread workgroup (16 waves). gates[64,2048] = h @ W.T + bias,
// C tiled as 128 groups of (16 rows x 16 cols-per-gate); each wave owns 8
// groups and computes i/f/g/o tiles for its columns so the cell update is
// register-local. h ping-pongs in 128KB dynamic LDS; c stays in global fp32.
__global__ void k_lstm(const _Float16* __restrict__ Whh,
                       const _Float16* __restrict__ Wcomb,
                       const float* __restrict__ bias0,
                       const float* __restrict__ biasC,
                       const _Float16* __restrict__ h0,
                       float* __restrict__ c_state,
                       _Float16* __restrict__ outs) {
  extern __shared__ _Float16 hbuf[];            // [2][64][512]
  const int tid = threadIdx.x;
  const int wave = tid >> 5, lane = tid & 31;
  const int n16 = lane & 15, hi = lane >> 4;

  for (int i = tid; i < 64 * 512; i += 512) hbuf[i] = h0[i];
  __syncthreads();

  for (int t = 0; t < 20; ++t) {
    const _Float16* W   = (t == 0) ? Whh : Wcomb;
    const float* bias   = (t == 0) ? bias0 : biasC;
    const _Float16* cur = hbuf + (t & 1) * (64 * 512);
    _Float16* nxt       = hbuf + ((t + 1) & 1) * (64 * 512);

    for (int g = 0; g < 8; ++g) {
      int G = wave + 16 * g;                    // 0..127
      int r0 = (G & 3) * 16;                    // batch tile
      int col0 = (G >> 2) * 16;                 // hidden column tile
      v8f acc[4];
#pragma unroll
      for (int q = 0; q < 4; ++q) {
        float b = bias[q * 512 + col0 + n16];
#pragma unroll
        for (int j = 0; j < 8; ++j) acc[q][j] = b;
      }
      for (int kk = 0; kk < 512; kk += 32) {
        v16h a = frag_ld(cur, 512, r0, kk);     // LDS -> ds_load
#pragma unroll
        for (int q = 0; q < 4; ++q) {
          v16h bf = frag_ld(W, 512, q * 512 + col0, kk);
          acc[q] = wmma_f16(a, bf, acc[q]);
        }
      }
#pragma unroll
      for (int j = 0; j < 8; ++j) {
        int row  = r0 + hi * 8 + j;             // batch index
        int colh = col0 + n16;                  // hidden index
        float ig = sigf(acc[0][j]);
        float fg = sigf(acc[1][j]);
        float gg = tanhf(acc[2][j]);
        float og = sigf(acc[3][j]);
        float* cp = c_state + row * 512 + colh;
        float cn = fg * (*cp) + ig * gg;
        *cp = cn;
        float hn = og * tanhf(cn);
        _Float16 hh = (_Float16)hn;
        nxt[row * 512 + colh] = hh;
        outs[(size_t)(row * 20 + t) * 512 + colh] = hh;   // [b][t][512]
      }
    }
    __syncthreads();
  }
}

// --------------------------- FF / FC GEMMs ---------------------------------

__global__ void k_ff(const _Float16* __restrict__ outs, const _Float16* __restrict__ Wf,
                     const float* __restrict__ bf, _Float16* __restrict__ feat) {
  int gid = blockIdx.x * 8 + (threadIdx.x >> 5);
  if (gid >= 80 * 32) return;
  int lane = threadIdx.x & 31, n16 = lane & 15, hi = lane >> 4;
  int r0 = (gid >> 5) * 16, c0 = (gid & 31) * 16;
  float bi = bf[c0 + n16];
  v8f acc;
#pragma unroll
  for (int j = 0; j < 8; ++j) acc[j] = bi;
  for (int kk = 0; kk < 512; kk += 32)
    acc = wmma_f16(frag_ld(outs, 512, r0, kk), frag_ld(Wf, 512, c0, kk), acc);
#pragma unroll
  for (int j = 0; j < 8; ++j)
    feat[(size_t)(r0 + hi * 8 + j) * 640 + c0 + n16] = (_Float16)lrelu(acc[j]);
}

__global__ void k_fc(const _Float16* __restrict__ feat, const _Float16* __restrict__ W,
                     const float* __restrict__ b, _Float16* __restrict__ act0) {
  int gid = blockIdx.x * 8 + (threadIdx.x >> 5);
  if (gid >= 80 * 256) return;
  int lane = threadIdx.x & 31, n16 = lane & 15, hi = lane >> 4;
  int r0 = (gid >> 8) * 16, c0 = (gid & 255) * 16;
  float bi = b[c0 + n16];
  v8f acc;
#pragma unroll
  for (int j = 0; j < 8; ++j) acc[j] = bi;
  for (int kk = 0; kk < 640; kk += 32)
    acc = wmma_f16(frag_ld(feat, 640, r0, kk), frag_ld(W, 640, c0, kk), acc);
  int cc = c0 + n16;
  int ch = cc >> 4, y = (cc >> 2) & 3, x = cc & 3;     // reshape [.,256,4,4]
#pragma unroll
  for (int j = 0; j < 8; ++j) {
    int img = r0 + hi * 8 + j;
    act0[(((size_t)img * 6 + (y + 1)) * 6 + (x + 1)) * 256 + ch] = (_Float16)lrelu(acc[j]);
  }
}

// --------------------------- implicit-GEMM deconv --------------------------
// ConvTranspose(k4,s2,p1): output (oy,ox) parity class (py,px) is a GEMM with
// K = 4*Cin over padded NHWC input at ipy = ((oy+1)>>1)+ty, ipx = ((ox+1)>>1)+tx.
__global__ void k_deconv(const _Float16* __restrict__ act, int Hp, int Cin,
                         const _Float16* __restrict__ bp, int Cout,
                         int S, float* __restrict__ raw, int totTiles) {
  int gid = blockIdx.x * 8 + (threadIdx.x >> 5);
  if (gid >= totTiles) return;
  int lane = threadIdx.x & 31, n16 = lane & 15, hi = lane >> 4;
  int Sh = S >> 1, ptiles = (Sh * Sh) >> 4, nt = Cout >> 4, K = Cin << 2;
  int cot = gid % nt; int r = gid / nt;
  int pt  = r % ptiles; r /= ptiles;
  int par = r & 3; int img = r >> 2;
  int py = par >> 1, px = par & 1, co0 = cot << 4;
  int idx = pt * 16 + n16;
  int ay = idx / Sh, bx = idx % Sh;
  v8f acc;
#pragma unroll
  for (int j = 0; j < 8; ++j) acc[j] = 0.f;
  const _Float16* bpp = bp + (size_t)par * Cout * K;
  for (int kk = 0; kk < K; kk += 32) {
    int tap = kk / Cin, c0k = kk - tap * Cin;
    int ty = tap >> 1, tx = tap & 1;
    const _Float16* ap = act +
        (((size_t)img * Hp + (ay + py + ty)) * Hp + (bx + px + tx)) * Cin +
        c0k + ((lane >> 4) << 3);
    union { v16h v; struct { v8h a, b; } s; } u;
    u.s.a = *(const v8h*)ap;
    u.s.b = *(const v8h*)(ap + 16);
    acc = wmma_f16(u.v, frag_ld(bpp, K, co0, kk), acc);
  }
#pragma unroll
  for (int j = 0; j < 8; ++j) {
    int id2 = pt * 16 + hi * 8 + j;
    int ay2 = id2 / Sh, bx2 = id2 % Sh;
    raw[(((size_t)img * S + (2 * ay2 + py)) * S + (2 * bx2 + px)) * Cout + co0 + n16] = acc[j];
  }
}

// --------------------------- BatchNorm (training) --------------------------

__global__ void k_bnstats(const float* __restrict__ raw, int count, int C,
                          const float* __restrict__ g, const float* __restrict__ be,
                          float* __restrict__ scale, float* __restrict__ shift) {
  __shared__ float sh[512];
  int c = blockIdx.x, tid = threadIdx.x;
  float s = 0.f, s2 = 0.f;
  for (int i = tid; i < count; i += 256) {
    float v = raw[(size_t)i * C + c];
    s += v; s2 += v * v;
  }
  sh[tid] = s; sh[256 + tid] = s2;
  __syncthreads();
  for (int off = 128; off > 0; off >>= 1) {
    if (tid < off) { sh[tid] += sh[tid + off]; sh[256 + tid] += sh[256 + tid + off]; }
    __syncthreads();
  }
  if (tid == 0) {
    float mean = sh[0] / count;
    float var  = sh[256] / count - mean * mean;   // biased variance
    float sc   = rsqrtf(var + 1e-5f) * g[c];
    scale[c] = sc;
    shift[c] = be[c] - mean * sc;
  }
}

__global__ void k_bnapply(const float* __restrict__ raw, const float* __restrict__ sc,
                          const float* __restrict__ sf, _Float16* __restrict__ act,
                          int C, int S, int Hp, long total) {
  long idx = (long)blockIdx.x * blockDim.x + threadIdx.x;
  if (idx >= total) return;
  int c = (int)(idx % C);
  long p = idx / C;
  int ox = (int)(p % S); p /= S;
  int oy = (int)(p % S); long img = p / S;
  float v = lrelu(raw[idx] * sc[c] + sf[c]);
  act[(((size_t)img * Hp + (oy + 1)) * Hp + (ox + 1)) * C + c] = (_Float16)v;
}

// --------------------------- final deconv + sigmoid ------------------------
// Cout == 1 would waste 15/16 of a WMMA tile: do it as scalar fp32 (128 MACs
// per output pixel), which also keeps the sigmoid input in full precision.
__global__ void k_final(const _Float16* __restrict__ act3, const float* __restrict__ wf,
                        const float* __restrict__ bfp, float* __restrict__ out) {
  int idx = blockIdx.x * blockDim.x + threadIdx.x;
  if (idx >= 1280 * 4096) return;
  int n = idx >> 12, rem = idx & 4095;
  int oy = rem >> 6, ox = rem & 63;
  int py = oy & 1, px = ox & 1;
  int iy0 = (oy + 1) >> 1, ix0 = (ox + 1) >> 1;
  float s = bfp[0];
#pragma unroll
  for (int ty = 0; ty < 2; ++ty)
#pragma unroll
    for (int tx = 0; tx < 2; ++tx) {
      const _Float16* ap = act3 + (((size_t)n * 34 + (iy0 + ty)) * 34 + (ix0 + tx)) * 32;
      int wy = 3 - 2 * ty - py, wx = 3 - 2 * tx - px;
      const float* wp = wf + wy * 4 + wx;
#pragma unroll
      for (int ci = 0; ci < 32; ++ci) s += (float)ap[ci] * wp[ci * 16];
    }
  out[idx] = 1.0f / (1.0f + __expf(-s));   // flat == ((b*20+t)*64+oy)*64+ox
}

// --------------------------- host orchestration ----------------------------

extern "C" void kernel_launch(void* const* d_in, const int* in_sizes, int n_in,
                              void* d_out, int out_size, void* d_ws, size_t ws_size,
                              hipStream_t stream) {
  const float* z_content = (const float*)d_in[0];
  const float* z_motion  = (const float*)d_in[1];
  const float* W_h  = (const float*)d_in[2];
  const float* b_h  = (const float*)d_in[3];
  const float* W_c  = (const float*)d_in[4];
  const float* b_c  = (const float*)d_in[5];
  const float* W_ih = (const float*)d_in[6];
  const float* W_hh = (const float*)d_in[7];
  const float* b_ih = (const float*)d_in[8];
  const float* b_hh = (const float*)d_in[9];
  const float* W_f  = (const float*)d_in[10];
  const float* b_f  = (const float*)d_in[11];
  const float* init_in = (const float*)d_in[12];
  const float* fc_W = (const float*)d_in[13];
  const float* fc_b = (const float*)d_in[14];
  const float* w0 = (const float*)d_in[15];
  const float* g0 = (const float*)d_in[16];
  const float* be0 = (const float*)d_in[17];
  const float* w1 = (const float*)d_in[18];
  const float* g1 = (const float*)d_in[19];
  const float* be1 = (const float*)d_in[20];
  const float* w2 = (const float*)d_in[21];
  const float* g2 = (const float*)d_in[22];
  const float* be2 = (const float*)d_in[23];
  const float* wf = (const float*)d_in[24];
  const float* bf = (const float*)d_in[25];
  float* out = (float*)d_out;
  (void)in_sizes; (void)n_in; (void)out_size; (void)ws_size;

  char* ws = (char*)d_ws;
  size_t off = 0;
  auto alloc = [&](size_t bytes) -> char* {
    char* p = ws + off;
    off += (bytes + 255) & ~(size_t)255;
    return p;
  };
  _Float16* WhhH   = (_Float16*)alloc((size_t)2048 * 512 * 2);
  _Float16* WcombH = (_Float16*)alloc((size_t)2048 * 512 * 2);
  _Float16* WfH    = (_Float16*)alloc((size_t)512 * 512 * 2);
  _Float16* fcWH   = (_Float16*)alloc((size_t)4096 * 640 * 2);
  _Float16* Bp0    = (_Float16*)alloc((size_t)4 * 128 * 1024 * 2);
  _Float16* Bp1    = (_Float16*)alloc((size_t)4 * 64 * 512 * 2);
  _Float16* Bp2    = (_Float16*)alloc((size_t)4 * 32 * 256 * 2);
  float* bias0     = (float*)alloc(2048 * 4);
  float* biasC     = (float*)alloc(2048 * 4);
  _Float16* h0     = (_Float16*)alloc((size_t)64 * 512 * 2);
  float* cst       = (float*)alloc((size_t)64 * 512 * 4);
  _Float16* outs   = (_Float16*)alloc((size_t)1280 * 512 * 2);
  _Float16* feat   = (_Float16*)alloc((size_t)1280 * 640 * 2);
  _Float16* act0   = (_Float16*)alloc((size_t)1280 * 36 * 256 * 2);    // 6x6x256 padded
  _Float16* act1   = (_Float16*)alloc((size_t)1280 * 100 * 128 * 2);   // 10x10x128
  _Float16* act2   = (_Float16*)alloc((size_t)1280 * 324 * 64 * 2);    // 18x18x64
  _Float16* act3   = (_Float16*)alloc((size_t)1280 * 1156 * 32 * 2);   // 34x34x32
  float* raw       = (float*)alloc((size_t)1280 * 1024 * 32 * 4);      // shared raw buf
  float* scale     = (float*)alloc(128 * 4);
  float* shift     = (float*)alloc(128 * 4);

  // zero all padded activation buffers (borders must be 0 every call)
  long actHalves = ((char*)raw - (char*)act0) / 2;
  k_zero_h<<<4096, 256, 0, stream>>>(act0, actHalves);

  // weight / bias preparation
  k_cvt_lstm<<<(2048 * 512 + 255) / 256, 256, 0, stream>>>(W_ih, W_hh, WhhH, WcombH);
  k_cvt<<<(512 * 512 + 255) / 256, 256, 0, stream>>>(W_f, WfH, 512 * 512);
  k_cvt<<<(4096 * 640 + 255) / 256, 256, 0, stream>>>(fc_W, fcWH, 4096 * 640);
  k_build_bp<<<(16 * 256 * 128 + 255) / 256, 256, 0, stream>>>(w0, Bp0, 256, 128);
  k_build_bp<<<(16 * 128 * 64 + 255) / 256, 256, 0, stream>>>(w1, Bp1, 128, 64);
  k_build_bp<<<(16 * 64 * 32 + 255) / 256, 256, 0, stream>>>(w2, Bp2, 64, 32);
  k_bias0<<<8, 256, 0, stream>>>(b_ih, b_hh, W_ih, init_in, bias0, biasC);
  k_h0c0<<<128, 256, 0, stream>>>(z_content, z_motion, W_h, b_h, W_c, b_c, h0, cst);

  // LSTM: one persistent workgroup, 128KB dynamic LDS ping-pong for h
  k_lstm<<<1, 512, 2 * 64 * 512 * 2, stream>>>(WhhH, WcombH, bias0, biasC, h0, cst, outs);

  // feature FF + content concat, then FC into padded NHWC act0
  k_ff<<<320, 256, 0, stream>>>(outs, WfH, b_f, feat);
  k_featzc<<<(1280 * 128 + 255) / 256, 256, 0, stream>>>(z_content, feat);
  k_fc<<<2560, 256, 0, stream>>>(feat, fcWH, fc_b, act0);

  // deconv0: 256->128, 4x4 -> 8x8
  k_deconv<<<5120, 256, 0, stream>>>(act0, 6, 256, Bp0, 128, 8, raw, 40960);
  k_bnstats<<<128, 256, 0, stream>>>(raw, 1280 * 64, 128, g0, be0, scale, shift);
  k_bnapply<<<40960, 256, 0, stream>>>(raw, scale, shift, act1, 128, 8, 10, 1280L * 64 * 128);

  // deconv1: 128->64, 8x8 -> 16x16
  k_deconv<<<10240, 256, 0, stream>>>(act1, 10, 128, Bp1, 64, 16, raw, 81920);
  k_bnstats<<<64, 256, 0, stream>>>(raw, 1280 * 256, 64, g1, be1, scale, shift);
  k_bnapply<<<81920, 256, 0, stream>>>(raw, scale, shift, act2, 64, 16, 18, 1280L * 256 * 64);

  // deconv2: 64->32, 16x16 -> 32x32
  k_deconv<<<20480, 256, 0, stream>>>(act2, 18, 64, Bp2, 32, 32, raw, 163840);
  k_bnstats<<<32, 256, 0, stream>>>(raw, 1280 * 1024, 32, g2, be2, scale, shift);
  k_bnapply<<<163840, 256, 0, stream>>>(raw, scale, shift, act3, 32, 32, 34, 1280L * 1024 * 32);

  // final 32->1 deconv + sigmoid -> d_out [64,20,1,64,64]
  k_final<<<20480, 256, 0, stream>>>(act3, wf, bf, out);
}
